// AttentionRNN_76295799046675
// MI455X (gfx1250) — matrix operand order repaired and checked by reading, not compiled, not run on hardware
//
#include <hip/hip_runtime.h>
#include <hip/hip_bf16.h>

// ---------------------------------------------------------------------------
// AttentionRNN for MI455X (gfx1250, wave32, WMMA).
// All GEMMs run as v_wmma_f32_16x16x32_f16 (f16 inputs, f32 accumulate).
// f16 tile staging uses GLOBAL_LOAD_ASYNC_TO_LDS_B128 (ASYNCcnt-tracked,
// guarded by __has_builtin), else falls back to ds_store staging.
// ---------------------------------------------------------------------------

typedef _Float16 f16;
typedef _Float16 f16x16 __attribute__((ext_vector_type(16)));
typedef _Float16 f16x8  __attribute__((ext_vector_type(8)));
typedef float    f32x8  __attribute__((ext_vector_type(8)));
typedef int      i32x4  __attribute__((ext_vector_type(4)));

#define B_  32
#define S_  512
#define E_  512
#define H_  512
#define NH_ 8
#define HD_ 64
#define MS_ (B_ * S_)   // 16384 rows for the projection GEMMs

// ---- optional async global->LDS copies (ASYNCcnt path) --------------------
#if defined(__has_builtin)
#  if __has_builtin(__builtin_amdgcn_global_load_async_to_lds_b128) && \
      __has_builtin(__builtin_amdgcn_s_wait_asynccnt)
#    define USE_ASYNC_COPY 1
#  endif
#endif
#ifndef USE_ASYNC_COPY
#  define USE_ASYNC_COPY 0
#endif

#if USE_ASYNC_COPY
typedef __attribute__((address_space(1))) i32x4 gbl_i32x4;   // global v4i
typedef __attribute__((address_space(3))) i32x4 lds_i32x4;   // LDS v4i
#endif

// copy one 16B chunk (8 halves) from global to LDS
__device__ __forceinline__ void copy16B_to_lds(f16* lds_dst, const f16* gsrc) {
#if USE_ASYNC_COPY
  // generic LDS pointers carry the LDS byte offset in the low 32 bits;
  // generic global pointers are the 64-bit VA. Reinterpret via uintptr.
  gbl_i32x4* g = (gbl_i32x4*)(unsigned long long)(const void*)gsrc;
  lds_i32x4* l = (lds_i32x4*)(unsigned long long)(void*)lds_dst;
  __builtin_amdgcn_global_load_async_to_lds_b128(g, l, 0, 0);
#else
  *(f16x8*)lds_dst = *(const f16x8*)gsrc;
#endif
}

// wait for staged tiles (async copies if used) + block barrier
__device__ __forceinline__ void wait_tiles() {
#if USE_ASYNC_COPY
  __builtin_amdgcn_s_wait_asynccnt(0);
#endif
  __syncthreads();
}

__device__ __forceinline__ f32x8 zero8() {
  f32x8 z;
#pragma unroll
  for (int i = 0; i < 8; ++i) z[i] = 0.0f;
  return z;
}

// A fragment (16x32 f16): lane -> m = lane%16 ; halves -> k = (lane/16)*8 + {0..7, 16..23}
__device__ __forceinline__ f16x16 ld_a_frag(const f16* base, int stride, int lane) {
  const f16* p = base + (lane & 15) * stride + (lane >> 4) * 8;
  f16x16 a;
  ((f16x8*)&a)[0] = *(const f16x8*)p;
  ((f16x8*)&a)[1] = *(const f16x8*)(p + 16);
  return a;
}

// B fragment (32x16 f16), stored as rows of n with k contiguous:
// lane -> n = lane%16 ; halves -> k = (lane/16)*8 + {0..7, 16..23}
__device__ __forceinline__ f16x16 ld_b_frag(const f16* base, int stride, int lane) {
  const f16* p = base + (lane & 15) * stride + (lane >> 4) * 8;
  f16x16 b;
  ((f16x8*)&b)[0] = *(const f16x8*)p;
  ((f16x8*)&b)[1] = *(const f16x8*)(p + 16);
  return b;
}

__device__ __forceinline__ f32x8 wmma_f16(f16x16 a, f16x16 b, f32x8 c) {
  return __builtin_amdgcn_wmma_f32_16x16x32_f16(false, a, false, b, (short)0, c,
                                                false, false);
}

// ---------------------------------------------------------------------------
// Generic NT GEMM:  C[m,n] = sum_k A[m,k] * B[n,k] + bias[n], C stored f16.
// Source precision is a template parameter -> branch-free inner loop.
// WG = 128 thr = 4 waves, tile M64 x N64, K chunks of 32.
// ---------------------------------------------------------------------------
template <int A_F16, int B_F16>
__global__ __launch_bounds__(128) void gemm_nt_kernel(
    const void* __restrict__ Ap, size_t lda,
    const void* __restrict__ Bp, size_t ldb,
    const float* __restrict__ bias,
    f16* __restrict__ C, size_t ldc, int K)
{
  __shared__ alignas(16) f16 As[64 * 48];
  __shared__ alignas(16) f16 Bs[64 * 48];
  const int tid  = threadIdx.x;
  const int lane = tid & 31;
  const int wv   = tid >> 5;
  const size_t mbase = (size_t)blockIdx.x * 64;
  const size_t nbase = (size_t)blockIdx.y * 64;

  f32x8 acc[4];
#pragma unroll
  for (int j = 0; j < 4; ++j) acc[j] = zero8();

  for (int k0 = 0; k0 < K; k0 += 32) {
    // ---- stage A tile (64 rows x 32 k) into LDS as f16 ----
    if (A_F16) {
      const f16* A = (const f16*)Ap;
#pragma unroll
      for (int i = 0; i < 2; ++i) {
        int c = tid + 128 * i;              // 256 chunks of 8 halves
        int r = c >> 2, col = (c & 3) * 8;
        copy16B_to_lds(&As[r * 48 + col], &A[(mbase + r) * lda + k0 + col]);
      }
    } else {
      const float* A = (const float*)Ap;
#pragma unroll
      for (int i = 0; i < 4; ++i) {
        int c = tid + 128 * i;              // 512 chunks of 4 floats
        int r = c >> 3, col = (c & 7) * 4;
        float4 f = *(const float4*)&A[(mbase + r) * lda + k0 + col];
        f16* d = &As[r * 48 + col];
        d[0] = (f16)f.x; d[1] = (f16)f.y; d[2] = (f16)f.z; d[3] = (f16)f.w;
      }
    }
    // ---- stage B tile (64 n-rows x 32 k) ----
    if (B_F16) {
      const f16* Bm = (const f16*)Bp;
#pragma unroll
      for (int i = 0; i < 2; ++i) {
        int c = tid + 128 * i;
        int r = c >> 2, col = (c & 3) * 8;
        copy16B_to_lds(&Bs[r * 48 + col], &Bm[(nbase + r) * ldb + k0 + col]);
      }
    } else {
      const float* Bm = (const float*)Bp;
#pragma unroll
      for (int i = 0; i < 4; ++i) {
        int c = tid + 128 * i;
        int r = c >> 3, col = (c & 7) * 4;
        float4 f = *(const float4*)&Bm[(nbase + r) * ldb + k0 + col];
        f16* d = &Bs[r * 48 + col];
        d[0] = (f16)f.x; d[1] = (f16)f.y; d[2] = (f16)f.z; d[3] = (f16)f.w;
      }
    }
    // prefetch next k-chunk of B weights (emits global_prefetch_b8)
    if (k0 + 32 < K) {
      const char* pf = B_F16
          ? (const char*)((const f16*)Bp + (nbase + (tid >> 2)) * ldb + k0 + 32)
          : (const char*)((const float*)Bp + (nbase + (tid >> 2)) * ldb + k0 + 32);
      __builtin_prefetch(pf, 0, 1);
    }
    wait_tiles();

    f16x16 af = ld_a_frag(&As[wv * 16 * 48], 48, lane);
#pragma unroll
    for (int j = 0; j < 4; ++j) {
      f16x16 bf = ld_b_frag(&Bs[j * 16 * 48], 48, lane);
      acc[j] = wmma_f16(af, bf, acc[j]);
    }
    __syncthreads();
  }

  const int nl   = lane & 15;
  const int msub = (lane >> 4) * 8;     // C layout: vgpr v -> m = (lane/16)*8 + v
#pragma unroll
  for (int j = 0; j < 4; ++j) {
    size_t n  = nbase + j * 16 + nl;
    float bv  = bias[n];
    size_t m0 = mbase + wv * 16 + msub;
#pragma unroll
    for (int v = 0; v < 8; ++v)
      C[(m0 + v) * ldc + n] = (f16)(acc[j][v] + bv);
  }
}

// ---------------------------------------------------------------------------
// Attention: one wave per (b, head, 16-row q tile).
// scores = (Q Kt)/8 -> softmax -> ctx = attn V. All matmuls via WMMA.
// ---------------------------------------------------------------------------
__global__ __launch_bounds__(32) void attn_kernel(
    const f16* __restrict__ Q, const f16* __restrict__ Kt,
    const f16* __restrict__ V, f16* __restrict__ Ctx)
{
  const int qt   = blockIdx.x;   // 0..31
  const int head = blockIdx.y;   // 0..7
  const int b    = blockIdx.z;   // 0..31
  const int lane = threadIdx.x;

  __shared__ alignas(16) f16   qs[16 * 80];    // Q tile  [q][d], stride 80
  __shared__ alignas(16) f16   ks[16 * 80];    // K tile  [key][d]
  __shared__ alignas(16) float sc[16 * 512];   // scores  [q][key]
  __shared__ alignas(16) f16   at[16 * 528];   // attn (A operand), stride 528
  __shared__ alignas(16) f16   vt[64 * 48];    // transposed V chunk [d][k]

  const size_t rowbase = (size_t)b * S_ + (size_t)qt * 16;
  const size_t hoff    = (size_t)head * HD_;

  // load Q tile: 16x64 halves = 128 chunks of 8
#pragma unroll
  for (int i = 0; i < 4; ++i) {
    int c = lane + 32 * i;
    int r = c >> 3, d = (c & 7) * 8;
    copy16B_to_lds(&qs[r * 80 + d], &Q[(rowbase + r) * E_ + hoff + d]);
  }

  // ---- scores = Q . K^T, per 16-key tile ----
  for (int kt = 0; kt < 32; ++kt) {
#pragma unroll
    for (int i = 0; i < 4; ++i) {
      int c = lane + 32 * i;
      int r = c >> 3, d = (c & 7) * 8;
      copy16B_to_lds(&ks[r * 80 + d],
                     &Kt[((size_t)b * S_ + kt * 16 + r) * E_ + hoff + d]);
    }
    wait_tiles();
    f32x8 a = zero8();
#pragma unroll
    for (int kc = 0; kc < 2; ++kc)      // HD=64 -> two K32 steps
      a = wmma_f16(ld_a_frag(qs + kc * 32, 80, lane),
                   ld_b_frag(ks + kc * 32, 80, lane), a);
    const int n  = lane & 15;
    const int ms = (lane >> 4) * 8;
#pragma unroll
    for (int v = 0; v < 8; ++v)
      sc[(ms + v) * 512 + kt * 16 + n] = a[v] * 0.125f;   // 1/sqrt(64)
    __syncthreads();
  }

  // ---- softmax rows (lanes 0..15 each own one q-row) ----
  if (lane < 16) {
    float* row = &sc[lane * 512];
    float mx = -3.0e38f;
    for (int k = 0; k < 512; ++k) mx = fmaxf(mx, row[k]);
    float s = 0.0f;
    for (int k = 0; k < 512; ++k) { float e = __expf(row[k] - mx); s += e; row[k] = e; }
    float inv = 1.0f / s;
    for (int k = 0; k < 512; ++k) at[lane * 528 + k] = (f16)(row[k] * inv);
  }
  __syncthreads();

  // ---- ctx = attn . V  (K = 512 over 16 chunks of 32) ----
  f32x8 acc[4];
#pragma unroll
  for (int j = 0; j < 4; ++j) acc[j] = zero8();

  for (int kc = 0; kc < 16; ++kc) {
    // load V chunk (32 k-rows x 64 d) transposed into vt[d][k]
    const f16* vr = &V[((size_t)b * S_ + kc * 32 + lane) * E_ + hoff];
#pragma unroll
    for (int dq = 0; dq < 8; ++dq) {
      f16x8 x = *(const f16x8*)&vr[dq * 8];
#pragma unroll
      for (int u = 0; u < 8; ++u) vt[(dq * 8 + u) * 48 + lane] = x[u];
    }
    __syncthreads();
    f16x16 af = ld_a_frag(at + kc * 32, 528, lane);
#pragma unroll
    for (int dt = 0; dt < 4; ++dt) {
      f16x16 bf = ld_b_frag(vt + dt * 16 * 48, 48, lane);
      acc[dt] = wmma_f16(af, bf, acc[dt]);
    }
    __syncthreads();
  }

  const int n  = lane & 15;
  const int ms = (lane >> 4) * 8;
#pragma unroll
  for (int dt = 0; dt < 4; ++dt)
#pragma unroll
    for (int v = 0; v < 8; ++v)
      Ctx[(rowbase + ms + v) * E_ + hoff + dt * 16 + n] = (f16)acc[dt][v];
}

// ---------------------------------------------------------------------------
// One RNN layer step: Hn = tanh(X @ Wih^T + Hp @ Whh^T + bih + bhh)
// Fused as a single K=1024 WMMA GEMM (M=32, N=512). Grid = 16 n-tiles of 32,
// WG = 4 waves in a 2(m) x 2(n) arrangement.
// ---------------------------------------------------------------------------
__global__ __launch_bounds__(128) void rnn_step_kernel(
    const f16* __restrict__ X, size_t xstride,
    const f16* __restrict__ Hp,
    const f16* __restrict__ Wih, const f16* __restrict__ Whh,
    const float* __restrict__ bih, const float* __restrict__ bhh,
    f16* __restrict__ Hn)
{
  __shared__ alignas(16) f16 As[32 * 48];
  __shared__ alignas(16) f16 Bs[32 * 48];
  const int tid  = threadIdx.x;
  const int lane = tid & 31;
  const int wv   = tid >> 5;
  const int mi   = wv & 1;     // m tile: 0..1 (M=32)
  const int ni   = wv >> 1;    // n tile: 0..1 within the 32-wide block
  const size_t nbase = (size_t)blockIdx.x * 32;

  f32x8 acc = zero8();

  for (int k0 = 0; k0 < 2 * H_; k0 += 32) {
    // A tile 32 x 32: 128 chunks of 8 halves, one per thread
    {
      int r = tid >> 2, col = (tid & 3) * 8;
      const f16* src = (k0 < H_)
          ? (X + (size_t)r * xstride + k0 + col)
          : (Hp + (size_t)r * H_ + (k0 - H_) + col);
      copy16B_to_lds(&As[r * 48 + col], src);
    }
    // B tile 32 x 32 from Wih / Whh (pre-converted f16, row-major [n][k])
    {
      int r = tid >> 2, col = (tid & 3) * 8;
      size_t n = nbase + r;
      const f16* src = (k0 < H_)
          ? (Wih + n * H_ + k0 + col)
          : (Whh + n * H_ + (k0 - H_) + col);
      copy16B_to_lds(&Bs[r * 48 + col], src);
    }
    wait_tiles();
    acc = wmma_f16(ld_a_frag(As + mi * 16 * 48, 48, lane),
                   ld_b_frag(Bs + ni * 16 * 48, 48, lane), acc);
    __syncthreads();
  }

  const int n    = (int)nbase + ni * 16 + (lane & 15);
  const float bs = bih[n] + bhh[n];
  const int m0   = mi * 16 + (lane >> 4) * 8;
#pragma unroll
  for (int v = 0; v < 8; ++v)
    Hn[(size_t)(m0 + v) * H_ + n] = (f16)tanhf(acc[v] + bs);
}

// ---------------------------------------------------------------------------
// Helpers
// ---------------------------------------------------------------------------
__global__ void cvt_f32_f16_kernel(f16* __restrict__ dst,
                                   const float* __restrict__ src, int n) {
  int i = blockIdx.x * blockDim.x + threadIdx.x;
  if (i < n) dst[i] = (f16)src[i];
}

__global__ void zero_f16_kernel(f16* __restrict__ p, int n) {
  int i = blockIdx.x * blockDim.x + threadIdx.x;
  if (i < n) p[i] = (f16)0.0f;
}

__global__ void fc_kernel(const f16* __restrict__ Hfin,
                          const float* __restrict__ Wfc,
                          const float* __restrict__ bfc,
                          float* __restrict__ out) {
  int b = threadIdx.x;          // 32 threads
  float s = 0.0f;
  for (int e = 0; e < H_; ++e) s += (float)Hfin[b * H_ + e] * Wfc[e];
  out[b] = s + bfc[0];
}

// ---------------------------------------------------------------------------
// Host orchestration
// ---------------------------------------------------------------------------
extern "C" void kernel_launch(void* const* d_in, const int* in_sizes, int n_in,
                              void* d_out, int out_size, void* d_ws, size_t ws_size,
                              hipStream_t stream) {
  (void)in_sizes; (void)n_in; (void)out_size; (void)ws_size;

  const float* x   = (const float*)d_in[0];
  const float* Wq  = (const float*)d_in[1];
  const float* bq  = (const float*)d_in[2];
  const float* Wk  = (const float*)d_in[3];
  const float* bk  = (const float*)d_in[4];
  const float* Wv  = (const float*)d_in[5];
  const float* bv  = (const float*)d_in[6];
  const float* Wo  = (const float*)d_in[7];
  const float* bo  = (const float*)d_in[8];
  const float* Wih = (const float*)d_in[9];
  const float* bih = (const float*)d_in[10];
  const float* Whh = (const float*)d_in[11];
  const float* bhh = (const float*)d_in[12];
  const float* Wfc = (const float*)d_in[13];
  const float* bfc = (const float*)d_in[14];

  // ---- scratch carve-up (256B aligned) ----
  char* ws = (char*)d_ws;
  size_t off = 0;
  auto alloc = [&](size_t bytes) -> void* {
    void* p = ws + off;
    off += (bytes + 255) & ~(size_t)255;
    return p;
  };
  const size_t MAT = (size_t)MS_ * E_ * sizeof(f16);   // 16.8 MB
  f16* q16   = (f16*)alloc(MAT);
  f16* k16   = (f16*)alloc(MAT);
  f16* v16   = (f16*)alloc(MAT);
  f16* ctx16 = (f16*)alloc(MAT);
  f16* Wih16 = (f16*)alloc((size_t)2 * H_ * H_ * sizeof(f16));
  f16* Whh16 = (f16*)alloc((size_t)2 * H_ * H_ * sizeof(f16));
  f16* hb0   = (f16*)alloc((size_t)2 * B_ * H_ * sizeof(f16));  // [layer][b][h]
  f16* hb1   = (f16*)alloc((size_t)2 * B_ * H_ * sizeof(f16));
  f16* att16 = q16;   // Q is dead after attention; reuse for attn output

  // ---- weight conversion + h0 = 0 ----
  {
    int n = 2 * H_ * H_;
    cvt_f32_f16_kernel<<<(n + 255) / 256, 256, 0, stream>>>(Wih16, Wih, n);
    cvt_f32_f16_kernel<<<(n + 255) / 256, 256, 0, stream>>>(Whh16, Whh, n);
    int nz = 2 * B_ * H_;
    zero_f16_kernel<<<(nz + 255) / 256, 256, 0, stream>>>(hb0, nz);
  }

  // ---- Q/K/V projections (f32 sources, inline convert) ----
  dim3 ggrid(MS_ / 64, E_ / 64);   // 256 x 8
  gemm_nt_kernel<0, 0><<<ggrid, 128, 0, stream>>>(x, (size_t)E_, Wq, (size_t)E_,
                                                  bq, q16, (size_t)E_, E_);
  gemm_nt_kernel<0, 0><<<ggrid, 128, 0, stream>>>(x, (size_t)E_, Wk, (size_t)E_,
                                                  bk, k16, (size_t)E_, E_);
  gemm_nt_kernel<0, 0><<<ggrid, 128, 0, stream>>>(x, (size_t)E_, Wv, (size_t)E_,
                                                  bv, v16, (size_t)E_, E_);

  // ---- attention ----
  attn_kernel<<<dim3(S_ / 16, NH_, B_), 32, 0, stream>>>(q16, k16, v16, ctx16);

  // ---- output projection (A is f16 ctx) ----
  gemm_nt_kernel<1, 0><<<ggrid, 128, 0, stream>>>(ctx16, (size_t)E_, Wo, (size_t)E_,
                                                  bo, att16, (size_t)E_, E_);

  // ---- stacked RNN scan over S (ping-pong h buffers) ----
  const size_t LOFF = (size_t)B_ * H_;      // per-layer offset in h buffers
  for (int t = 0; t < S_; ++t) {
    f16* src = (t & 1) ? hb1 : hb0;
    f16* dst = (t & 1) ? hb0 : hb1;
    // layer 0: input rows are att16[b, t, :]  (row stride S*E)
    rnn_step_kernel<<<16, 128, 0, stream>>>(
        att16 + (size_t)t * E_, (size_t)S_ * E_,
        src, Wih16, Whh16, bih, bhh, dst);
    // layer 1: input = layer-0's fresh output
    rnn_step_kernel<<<16, 128, 0, stream>>>(
        dst, (size_t)H_,
        src + LOFF, Wih16 + (size_t)H_ * H_, Whh16 + (size_t)H_ * H_,
        bih + H_, bhh + H_, dst + LOFF);
  }
  // t=511 is odd -> final state lives in hb0; layer-1 slice feeds the FC.
  fc_kernel<<<1, 32, 0, stream>>>(hb0 + LOFF, Wfc, bfc, (float*)d_out);
}
